// GemmaBlock_7206955123115
// MI455X (gfx1250) — matrix operand order repaired and checked by reading, not compiled, not run on hardware
//
#include <hip/hip_runtime.h>
#include <cstdint>
#include <cstddef>

// ---------------- problem constants (from reference) ----------------
#define B_ 2
#define L_ 2048
#define D_ 2560
#define H_ 8
#define KV_ 4
#define HD_ 256
#define FF_ 10240
#define WINDOW_ 1024
#define EPS_ 1e-6f
#define M_TOK (B_ * L_)          // 4096 token rows
#define NQ_ (H_ * HD_)           // 2048
#define NKV_ (KV_ * HD_)         // 1024
#define LOG_ROPE_BASE 9.210340371976184f  // ln(10000)

typedef unsigned int uint32;
typedef __attribute__((ext_vector_type(16))) __bf16 v16bf;
typedef __attribute__((ext_vector_type(8)))  float  v8f;

// 16-byte int vector type matching the async-LDS builtin's parameter type
typedef int v4i_vs __attribute__((__vector_size__(16)));
typedef __attribute__((address_space(1))) v4i_vs* gptr_v4i;
typedef __attribute__((address_space(3))) v4i_vs* lptr_v4i;

// ---- async global->LDS (probe-verified on this toolchain) ----
#if defined(__has_builtin)
#if __has_builtin(__builtin_amdgcn_global_load_async_to_lds_b128)
#define HAVE_ASYNC_LDS 1
#endif
#if __has_builtin(__builtin_amdgcn_s_wait_asynccnt)
#define HAVE_WAIT_ASYNC_BUILTIN 1
#endif
#endif

__device__ inline void wait_asynccnt0() {
#if defined(HAVE_WAIT_ASYNC_BUILTIN)
  __builtin_amdgcn_s_wait_asynccnt(0);
#else
  asm volatile("s_wait_asynccnt 0" ::: "memory");
#endif
}

// ---------------- helpers ----------------
__device__ inline unsigned short f2bf(float f) {
  uint32 u = __float_as_uint(f);
  u += 0x7FFFu + ((u >> 16) & 1u);   // round-to-nearest-even
  return (unsigned short)(u >> 16);
}
__device__ inline uint32 pack_bf16(float a, float b) {
  return (uint32)f2bf(a) | ((uint32)f2bf(b) << 16);
}

// block = 256 threads (8 wave32). Leading barrier protects shared-buffer reuse.
__device__ inline float block_reduce_sum256(float v, float* sred) {
  #pragma unroll
  for (int o = 16; o > 0; o >>= 1) v += __shfl_xor(v, o, 32);
  __syncthreads();
  if ((threadIdx.x & 31) == 0) sred[threadIdx.x >> 5] = v;
  __syncthreads();
  float s = 0.f;
  #pragma unroll
  for (int i = 0; i < 8; ++i) s += sred[i];
  return s;
}

// ---------------- weight convert + transpose: Wt[n*K + k] = bf16(W[k*N + n]) ----------------
__global__ __launch_bounds__(256) void conv_tr_kernel(const float* __restrict__ W,
                                                      unsigned short* __restrict__ Wt,
                                                      int K, int N) {
  size_t total = (size_t)K * (size_t)N;
  size_t stride = (size_t)gridDim.x * blockDim.x;
  for (size_t idx = (size_t)blockIdx.x * blockDim.x + threadIdx.x; idx < total; idx += stride) {
    size_t k = idx / (size_t)N;
    size_t n = idx - k * (size_t)N;
    Wt[n * (size_t)K + k] = f2bf(W[idx]);   // coalesced read
  }
}

// ---------------- rmsnorm over rows (one block per row); scale may be null; in-place safe ----------------
__global__ __launch_bounds__(256) void rmsnorm_kernel(const float* __restrict__ x,
                                                      const float* __restrict__ scale,
                                                      float* __restrict__ out, int ncols) {
  __shared__ float sred[8];
  size_t base = (size_t)blockIdx.x * ncols;
  float ss = 0.f;
  for (int c = threadIdx.x; c < ncols; c += 256) { float v = x[base + c]; ss += v * v; }
  float tot = block_reduce_sum256(ss, sred);
  float inv = rsqrtf(tot / (float)ncols + EPS_);
  for (int c = threadIdx.x; c < ncols; c += 256) {
    float v = x[base + c] * inv;
    if (scale) v *= scale[c];
    out[base + c] = v;
  }
}

// out = resid + rmsnorm(y, scale)
__global__ __launch_bounds__(256) void resid_rms_add_kernel(const float* __restrict__ resid,
                                                            const float* __restrict__ y,
                                                            const float* __restrict__ scale,
                                                            float* __restrict__ out, int ncols) {
  __shared__ float sred[8];
  size_t base = (size_t)blockIdx.x * ncols;
  float ss = 0.f;
  for (int c = threadIdx.x; c < ncols; c += 256) { float v = y[base + c]; ss += v * v; }
  float tot = block_reduce_sum256(ss, sred);
  float inv = rsqrtf(tot / (float)ncols + EPS_);
  for (int c = threadIdx.x; c < ncols; c += 256) {
    out[base + c] = resid[base + c] + y[base + c] * inv * scale[c];
  }
}

// ---------------- per-head rmsnorm + RoPE, fp32 in -> bf16 out. grid = M_TOK*nheads ----------------
__global__ __launch_bounds__(256) void rms_rope_kernel(const float* __restrict__ qk,
                                                       unsigned short* __restrict__ qkbf,
                                                       const float* __restrict__ scale,
                                                       const int* __restrict__ positions,
                                                       int nheads) {
  __shared__ float sred[8];
  __shared__ float sx[HD_];
  int bid = blockIdx.x;
  int m = bid / nheads;                       // token row (b*L + l)
  const float* p = qk + (size_t)bid * HD_;
  unsigned short* o = qkbf + (size_t)bid * HD_;
  int t = threadIdx.x;
  float v = p[t];
  float tot = block_reduce_sum256(v * v, sred);
  float inv = rsqrtf(tot / (float)HD_ + EPS_);
  sx[t] = v * inv * scale[t];
  __syncthreads();
  float pos = (float)positions[m];
  if (t < HD_ / 2) {
    float inv_ts = __expf(-(float)t * (LOG_ROPE_BASE / (float)(HD_ / 2)));
    float ang = pos * inv_ts;
    float s, c; __sincosf(ang, &s, &c);
    float x1 = sx[t], x2 = sx[t + HD_ / 2];
    o[t]           = f2bf(x1 * c - x2 * s);
    o[t + HD_ / 2] = f2bf(x2 * c + x1 * s);
  }
}

// ---------------- v: rmsnorm (no scale) then write transposed bf16 [b][kvh][hd][L] --------------
__global__ __launch_bounds__(256) void v_norm_tr_kernel(const float* __restrict__ vb,
                                                        unsigned short* __restrict__ vt) {
  __shared__ float sred[8];
  int bid = blockIdx.x;                       // = m*KV + kvh
  int m = bid / KV_, kvh = bid % KV_;
  int b = m / L_, l = m % L_;
  const float* p = vb + (size_t)bid * HD_;
  int t = threadIdx.x;
  float v = p[t];
  float tot = block_reduce_sum256(v * v, sred);
  float inv = rsqrtf(tot / (float)HD_ + EPS_);
  vt[((size_t)(b * KV_ + kvh) * HD_ + t) * L_ + l] = f2bf(v * inv);
}

// ---------------- WMMA bf16 GEMM: C[M,N] = A[M,K](fp32) * W[K,N], W given as Bt[N][K] bf16 ----------
// block 256 thr = 8 waves; block tile 64(M) x 128(N); BK = 64. 8 wmma per wave per K-iter.
__global__ __launch_bounds__(256) void gemm_bf16_kernel(const float* __restrict__ A,
                                                        const unsigned short* __restrict__ Bt,
                                                        float* __restrict__ C,
                                                        int K, int N) {
  __shared__ __align__(16) uint32 lds_a[64 * 32];    // 64 rows x 64 bf16
  __shared__ __align__(16) uint32 lds_b[128 * 32];   // 128 n-rows x 64 bf16 (K-contiguous)

  const int t = threadIdx.x;
  const int lane = t & 31;
  const int wave = t >> 5;
  const int wm = wave >> 2;            // 0..1 -> rows 32*wm
  const int wn = wave & 3;             // 0..3 -> cols 32*wn
  const int m_base = blockIdx.x * 64;
  const int n_base = blockIdx.y * 128;
  const int mrow = lane & 15;
  const int khalf = lane >> 4;

  v8f acc00 = {}, acc01 = {}, acc10 = {}, acc11 = {};

  for (int k0 = 0; k0 < K; k0 += 64) {
    __syncthreads();
    // stage A (64x64 fp32 -> bf16): 1024 uint2 stores, 4 per thread
    #pragma unroll
    for (int r = 0; r < 4; ++r) {
      int qd = t + r * 256;
      int row = qd >> 4, seg = qd & 15;           // seg = 4-float chunk
      const float* ap = A + (size_t)(m_base + row) * K + k0 + (seg << 2);
      float4 f4 = *(const float4*)ap;
      uint2 pk;
      pk.x = pack_bf16(f4.x, f4.y);
      pk.y = pack_bf16(f4.z, f4.w);
      *(uint2*)&lds_a[(row << 5) + (seg << 1)] = pk;
    }
    // stage B (128x64 bf16): 1024 x 16B async transfers, 4 per thread
    #pragma unroll
    for (int r = 0; r < 4; ++r) {
      int qd = t + r * 256;
      int nrow = qd >> 3, seg = qd & 7;           // seg = 8-bf16 chunk
      const unsigned short* bp = Bt + (size_t)(n_base + nrow) * K + k0 + (seg << 3);
      uint32* lp = &lds_b[(nrow << 5) + (seg << 2)];
#if defined(HAVE_ASYNC_LDS)
      __builtin_amdgcn_global_load_async_to_lds_b128(
          (gptr_v4i)(unsigned short*)bp,
          (lptr_v4i)lp, 0, 0);
#else
      *(uint4*)lp = *(const uint4*)bp;
#endif
    }
#if defined(HAVE_ASYNC_LDS)
    wait_asynccnt0();
#endif
    __syncthreads();

    #pragma unroll
    for (int ks = 0; ks < 2; ++ks) {
      union { v16bf v; uint32 u[8]; } fa0, fa1, fb0, fb1;
      const int ar0 = (wm << 5) + mrow;
      const int br0 = (wn << 5) + mrow;
      #pragma unroll
      for (int j = 0; j < 8; ++j) {
        int kk = (khalf << 4) + (j << 1);
        int kd = (ks << 4) + (kk >> 1);
        fa0.u[j] = lds_a[ar0 * 32 + kd];
        fa1.u[j] = lds_a[(ar0 + 16) * 32 + kd];
        int bd = (ks << 4) + (khalf << 3) + j;
        fb0.u[j] = lds_b[br0 * 32 + bd];
        fb1.u[j] = lds_b[(br0 + 16) * 32 + bd];
      }
      acc00 = __builtin_amdgcn_wmma_f32_16x16x32_bf16(false, fa0.v, false, fb0.v, (short)0, acc00, false, false);
      acc01 = __builtin_amdgcn_wmma_f32_16x16x32_bf16(false, fa0.v, false, fb1.v, (short)0, acc01, false, false);
      acc10 = __builtin_amdgcn_wmma_f32_16x16x32_bf16(false, fa1.v, false, fb0.v, (short)0, acc10, false, false);
      acc11 = __builtin_amdgcn_wmma_f32_16x16x32_bf16(false, fa1.v, false, fb1.v, (short)0, acc11, false, false);
    }
  }

  const int ng0 = n_base + (wn << 5) + mrow;
  const int mg0 = m_base + (wm << 5) + (khalf << 3);
  #pragma unroll
  for (int r = 0; r < 8; ++r) {
    C[(size_t)(mg0 + r) * N + ng0]           = acc00[r];
    C[(size_t)(mg0 + r) * N + ng0 + 16]      = acc01[r];
    C[(size_t)(mg0 + r + 16) * N + ng0]      = acc10[r];
    C[(size_t)(mg0 + r + 16) * N + ng0 + 16] = acc11[r];
  }
}

// ---------------- WMMA flash attention (windowed causal GQA) --------------------
// One wave per (16-query tile, head). block = 256 (8 independent waves).
// qbf: [M][H][HD] bf16 ; kbf: [M][KV][HD] bf16 ; vt: [b][kvh][HD][L] bf16 ; out fp32 [M][H][HD].
__global__ __launch_bounds__(256) void attn_wmma_kernel(const unsigned short* __restrict__ qbf,
                                                        const unsigned short* __restrict__ kbf,
                                                        const unsigned short* __restrict__ vt,
                                                        float* __restrict__ out) {
  __shared__ __align__(16) uint32 q_lds[8][16 * 128];  // per wave: 16 rows x 256 bf16
  __shared__ __align__(16) uint32 p_lds[8][16 * 16];   // per wave: 16 rows x 32 bf16

  const int t = threadIdx.x;
  const int lane = t & 31;
  const int wave = t >> 5;
  const int tile = blockIdx.x * 8 + wave;    // q-tile id in [0, (M/16)*H)
  const int qbase = (tile / H_) * 16;        // global token row of query 0
  const int h = tile % H_;
  const int b = qbase / L_;
  const int l0 = qbase % L_;                 // local position of query row 0
  const int kvh = h / (H_ / KV_);
  const int mrow = lane & 15;
  const int khalf = lane >> 4;

  // stage Q tile (16 x 256 bf16 = 2048 dwords) into this wave's LDS region
  for (int idx = lane; idx < 2048; idx += 32) {
    int row = idx >> 7, d = idx & 127;
    q_lds[wave][idx] = *(const uint32*)(qbf + ((size_t)(qbase + row) * H_ + h) * HD_ + (d << 1));
  }

  v8f acc[16];
  #pragma unroll
  for (int n = 0; n < 16; ++n) acc[n] = {};
  float rm[8], rs[8];
  #pragma unroll
  for (int r = 0; r < 8; ++r) { rm[r] = -3.0e38f; rs[r] = 0.f; }

  int c0 = l0 - (WINDOW_ - 1); if (c0 < 0) c0 = 0; c0 &= ~31;
  int c1 = (l0 + 15) & ~31;

  for (int c = c0; c <= c1; c += 32) {
    // ---- S = Q x K^T for 32 keys (two 16-key subtiles), accumulated over 256 k-dims ----
    v8f s0 = {}, s1 = {};
    #pragma unroll
    for (int kt = 0; kt < 2; ++kt) {
      v8f s = {};
      const unsigned short* kbase =
          kbf + ((size_t)(b * L_ + c + kt * 16 + mrow) * KV_ + kvh) * HD_;
      #pragma unroll
      for (int ks2 = 0; ks2 < 8; ++ks2) {
        union { v16bf v; uint32 u[8]; } fa, fb;
        #pragma unroll
        for (int j = 0; j < 8; ++j) {
          int kk = (ks2 << 5) + (khalf << 4) + (j << 1);
          fa.u[j] = q_lds[wave][(mrow << 7) + (kk >> 1)];
          int bk = (ks2 << 5) + (khalf << 4) + (j << 1);
          fb.u[j] = *(const uint32*)(kbase + bk);
        }
        s = __builtin_amdgcn_wmma_f32_16x16x32_bf16(false, fa.v, false, fb.v, (short)0, s, false, false);
      }
      if (kt == 0) s0 = s; else s1 = s;
    }

    // ---- masking + online softmax on S (D layout: vgpr r -> row 8*khalf+r, lane&15 -> key) ----
    const int key0 = c + mrow;
    const int key1 = key0 + 16;
    #pragma unroll
    for (int r = 0; r < 8; ++r) {
      const int qrow = l0 + (khalf << 3) + r;
      float a0 = (key0 <= qrow && key0 > qrow - WINDOW_) ? s0[r] : -10000.f;
      float a1 = (key1 <= qrow && key1 > qrow - WINDOW_) ? s1[r] : -10000.f;
      float mv = fmaxf(a0, a1);
      #pragma unroll
      for (int o = 8; o > 0; o >>= 1) mv = fmaxf(mv, __shfl_xor(mv, o, 16));
      const float nm = fmaxf(rm[r], mv);
      const float corr = __expf(rm[r] - nm);
      const float p0 = __expf(a0 - nm);
      const float p1 = __expf(a1 - nm);
      float psum = p0 + p1;
      #pragma unroll
      for (int o = 8; o > 0; o >>= 1) psum += __shfl_xor(psum, o, 16);
      rs[r] = rs[r] * corr + psum;
      rm[r] = nm;
      #pragma unroll
      for (int n = 0; n < 16; ++n) acc[n][r] *= corr;
      // write P into A-fragment source layout: [row][32 keys] bf16
      unsigned short* pp = (unsigned short*)&p_lds[wave][((khalf << 3) + r) << 4];
      pp[mrow] = f2bf(p0);
      pp[mrow + 16] = f2bf(p1);
    }

    // ---- O += P x V (A = P 16x32, B = V 32keys x 16hd per n-tile from transposed V) ----
    union { v16bf v; uint32 u[8]; } pa;
    #pragma unroll
    for (int j = 0; j < 8; ++j) {
      int kk = (khalf << 4) + (j << 1);
      pa.u[j] = p_lds[wave][(mrow << 4) + (kk >> 1)];
    }
    const unsigned short* vbase = vt + ((size_t)(b * KV_ + kvh) * HD_) * L_;
    #pragma unroll
    for (int n = 0; n < 16; ++n) {
      union { v16bf v; uint32 u[8]; } fb;
      const unsigned short* vrow = vbase + (size_t)(n * 16 + mrow) * L_ + c;
      #pragma unroll
      for (int j = 0; j < 8; ++j) fb.u[j] = *(const uint32*)(vrow + (khalf << 4) + (j << 1));
      acc[n] = __builtin_amdgcn_wmma_f32_16x16x32_bf16(false, pa.v, false, fb.v, (short)0, acc[n], false, false);
    }
  }

  // ---- normalize and store ----
  #pragma unroll
  for (int n = 0; n < 16; ++n) {
    #pragma unroll
    for (int r = 0; r < 8; ++r) {
      out[((size_t)(qbase + (khalf << 3) + r) * H_ + h) * HD_ + n * 16 + mrow] = acc[n][r] / rs[r];
    }
  }
}

// ---------------- hidden = gelu_exact(gate) * up, in place into gate ----------------
__global__ __launch_bounds__(256) void gelu_mul_kernel(float* __restrict__ g,
                                                       const float* __restrict__ u, size_t n) {
  size_t stride = (size_t)gridDim.x * blockDim.x;
  for (size_t idx = (size_t)blockIdx.x * blockDim.x + threadIdx.x; idx < n; idx += stride) {
    float x = g[idx];
    float ge = 0.5f * x * (1.0f + erff(x * 0.70710678118654752f));
    g[idx] = ge * u[idx];
  }
}

// ---------------- host orchestration ----------------
extern "C" void kernel_launch(void* const* d_in, const int* in_sizes, int n_in,
                              void* d_out, int out_size, void* d_ws, size_t ws_size,
                              hipStream_t stream) {
  const float* x              = (const float*)d_in[0];
  const int*   positions      = (const int*)  d_in[1];
  const float* wq             = (const float*)d_in[2];
  const float* wk             = (const float*)d_in[3];
  const float* wv             = (const float*)d_in[4];
  const float* wo             = (const float*)d_in[5];
  const float* q_scale        = (const float*)d_in[6];
  const float* k_scale        = (const float*)d_in[7];
  const float* pre_attn_scale = (const float*)d_in[8];
  const float* post_attn_scale= (const float*)d_in[9];
  const float* pre_ffw_scale  = (const float*)d_in[10];
  const float* post_ffw_scale = (const float*)d_in[11];
  const float* w_gate         = (const float*)d_in[12];
  const float* w_up           = (const float*)d_in[13];
  const float* w_down         = (const float*)d_in[14];
  float* outp = (float*)d_out;

  // workspace carve (256B aligned)
  char* wsb = (char*)d_ws;
  size_t off = 0;
  auto carve = [&](size_t bytes) -> char* {
    char* p = wsb + off;
    off += (bytes + 255) & ~(size_t)255;
    return p;
  };
  unsigned short* wq_t = (unsigned short*)carve((size_t)D_ * NQ_ * 2);
  unsigned short* wk_t = (unsigned short*)carve((size_t)D_ * NKV_ * 2);
  unsigned short* wv_t = (unsigned short*)carve((size_t)D_ * NKV_ * 2);
  unsigned short* wo_t = (unsigned short*)carve((size_t)NQ_ * D_ * 2);
  unsigned short* wg_t = (unsigned short*)carve((size_t)D_ * FF_ * 2);
  unsigned short* wu_t = (unsigned short*)carve((size_t)D_ * FF_ * 2);
  unsigned short* wd_t = (unsigned short*)carve((size_t)FF_ * D_ * 2);
  float* h        = (float*)carve((size_t)M_TOK * D_ * 4);
  float* qb       = (float*)carve((size_t)M_TOK * NQ_ * 4);
  float* kb       = (float*)carve((size_t)M_TOK * NKV_ * 4);
  float* vb       = (float*)carve((size_t)M_TOK * NKV_ * 4);
  unsigned short* qbf = (unsigned short*)carve((size_t)M_TOK * NQ_ * 2);
  unsigned short* kbf = (unsigned short*)carve((size_t)M_TOK * NKV_ * 2);
  unsigned short* vtb = (unsigned short*)carve((size_t)B_ * KV_ * HD_ * L_ * 2);
  float* attn_out = (float*)carve((size_t)M_TOK * NQ_ * 4);
  float* t0       = (float*)carve((size_t)M_TOK * D_ * 4);
  float* x1       = (float*)carve((size_t)M_TOK * D_ * 4);
  float* gate     = (float*)carve((size_t)M_TOK * FF_ * 4);
  float* up       = (float*)carve((size_t)M_TOK * FF_ * 4);
  (void)ws_size; (void)in_sizes; (void)n_in; (void)out_size;

  // 1) weights -> bf16, transposed to [N][K]
  conv_tr_kernel<<<8192, 256, 0, stream>>>(wq,     wq_t, D_,  NQ_);
  conv_tr_kernel<<<8192, 256, 0, stream>>>(wk,     wk_t, D_,  NKV_);
  conv_tr_kernel<<<8192, 256, 0, stream>>>(wv,     wv_t, D_,  NKV_);
  conv_tr_kernel<<<8192, 256, 0, stream>>>(wo,     wo_t, NQ_, D_);
  conv_tr_kernel<<<8192, 256, 0, stream>>>(w_gate, wg_t, D_,  FF_);
  conv_tr_kernel<<<8192, 256, 0, stream>>>(w_up,   wu_t, D_,  FF_);
  conv_tr_kernel<<<8192, 256, 0, stream>>>(w_down, wd_t, FF_, D_);

  // 2) pre-attention rmsnorm
  rmsnorm_kernel<<<M_TOK, 256, 0, stream>>>(x, pre_attn_scale, h, D_);

  // 3) q/k/v projections (WMMA bf16)
  gemm_bf16_kernel<<<dim3(M_TOK / 64, NQ_  / 128), 256, 0, stream>>>(h, wq_t, qb, D_, NQ_);
  gemm_bf16_kernel<<<dim3(M_TOK / 64, NKV_ / 128), 256, 0, stream>>>(h, wk_t, kb, D_, NKV_);
  gemm_bf16_kernel<<<dim3(M_TOK / 64, NKV_ / 128), 256, 0, stream>>>(h, wv_t, vb, D_, NKV_);

  // 4) per-head rmsnorm + rope -> bf16 (q,k); rmsnorm + transpose -> bf16 (v)
  rms_rope_kernel<<<M_TOK * H_,  256, 0, stream>>>(qb, qbf, q_scale, positions, H_);
  rms_rope_kernel<<<M_TOK * KV_, 256, 0, stream>>>(kb, kbf, k_scale, positions, KV_);
  v_norm_tr_kernel<<<M_TOK * KV_, 256, 0, stream>>>(vb, vtb);

  // 5) windowed attention (WMMA flash)
  attn_wmma_kernel<<<(M_TOK / 16) * H_ / 8, 256, 0, stream>>>(qbf, kbf, vtb, attn_out);

  // 6) output projection + residual rmsnorm add
  gemm_bf16_kernel<<<dim3(M_TOK / 64, D_ / 128), 256, 0, stream>>>(attn_out, wo_t, t0, NQ_, D_);
  resid_rms_add_kernel<<<M_TOK, 256, 0, stream>>>(x, t0, post_attn_scale, x1, D_);

  // 7) FFN
  rmsnorm_kernel<<<M_TOK, 256, 0, stream>>>(x1, pre_ffw_scale, h, D_);
  gemm_bf16_kernel<<<dim3(M_TOK / 64, FF_ / 128), 256, 0, stream>>>(h, wg_t, gate, D_, FF_);
  gemm_bf16_kernel<<<dim3(M_TOK / 64, FF_ / 128), 256, 0, stream>>>(h, wu_t, up,   D_, FF_);
  gelu_mul_kernel<<<8192, 256, 0, stream>>>(gate, up, (size_t)M_TOK * FF_);
  gemm_bf16_kernel<<<dim3(M_TOK / 64, D_ / 128), 256, 0, stream>>>(gate, wd_t, t0, FF_, D_);

  // 8) final residual rmsnorm add -> output
  resid_rms_add_kernel<<<M_TOK, 256, 0, stream>>>(x1, t0, post_ffw_scale, outp, D_);
}
